// Block_44427141710500
// MI455X (gfx1250) — matrix operand order repaired and checked by reading, not compiled, not run on hardware
//
#include <hip/hip_runtime.h>

typedef __bf16 bf16;
typedef __attribute__((ext_vector_type(16))) __bf16 v16bf;
typedef __attribute__((ext_vector_type(8)))  __bf16 v8bf;
typedef __attribute__((ext_vector_type(8)))  float  v8f;
typedef __attribute__((ext_vector_type(4)))  int    v4i;

#define WMMA_BF16(a,b,c) __builtin_amdgcn_wmma_f32_16x16x32_bf16(false,(a),false,(b),(short)0,(c),false,false)

// ---- async global->LDS copy (CDNA5 path, ASYNCcnt), with sync fallback ----
#if __has_builtin(__builtin_amdgcn_global_load_async_to_lds_b128) && __has_builtin(__builtin_amdgcn_s_wait_asynccnt)
#define HAVE_ASYNC 1
#define ASYNC_WAIT(n) __builtin_amdgcn_s_wait_asynccnt(n)
typedef __attribute__((address_space(1))) v4i* gas_v4i_p;
typedef __attribute__((address_space(3))) v4i* lds_v4i_p;
#else
#define HAVE_ASYNC 0
#define ASYNC_WAIT(n)
#endif

__device__ __forceinline__ void cp16_g2s(bf16* lds_dst, const bf16* gsrc){
#if HAVE_ASYNC
  __builtin_amdgcn_global_load_async_to_lds_b128(
      (gas_v4i_p)(void*)gsrc, (lds_v4i_p)(void*)lds_dst, 0, 0);
#else
  *(v8bf*)lds_dst = *(const v8bf*)gsrc;
#endif
}

// ---- model constants ----
#define BB 2
#define SS 2048
#define DD 1024
#define TT 4096          // B*S tokens
#define LQQ 768
#define LKV 512
#define HH 16
#define HD 64
#define EE 8
#define NSLOT 8192       // T*K routed slots
#define SLOTPAD 8320     // +128 tile tail padding (BM=128)

// ===================== WMMA fragment loaders (LDS, bf16) =====================
__device__ __forceinline__ v16bf frag_a(const bf16* p, int row_base, int stride, int kbase){
  int lane = threadIdx.x & 31;
  const bf16* pr = p + (size_t)(row_base + (lane & 15)) * stride + kbase + ((lane & 16) ? 8 : 0);
  v8bf lo = *(const v8bf*)pr;
  v8bf hi = *(const v8bf*)(pr + 16);
  return __builtin_shufflevector(lo,hi,0,1,2,3,4,5,6,7,8,9,10,11,12,13,14,15);
}
__device__ __forceinline__ v16bf frag_b(const bf16* p, int col_base, int stride, int kbase){
  int lane = threadIdx.x & 31;
  const bf16* pc = p + (size_t)(col_base + (lane & 15)) * stride + kbase + ((lane & 16) ? 8 : 0);
  v8bf lo = *(const v8bf*)pc;
  v8bf hi = *(const v8bf*)(pc + 16);
  return __builtin_shufflevector(lo,hi,0,1,2,3,4,5,6,7,8,9,10,11,12,13,14,15);
}

// ===================== transpose + fp32->bf16 convert: W[K][N] -> WT[N][K] =====================
__global__ __launch_bounds__(256) void transpose_cvt(const float* __restrict__ W, bf16* __restrict__ WT,
                                                     int K, int N){
  __shared__ float tile[32][33];
  int kb = blockIdx.y * 32, nb = blockIdx.x * 32;
  int tx = threadIdx.x & 31, ty = threadIdx.x >> 5;
  #pragma unroll
  for (int i = 0; i < 32; i += 8){
    int k = kb + ty + i, n = nb + tx;
    tile[ty + i][tx] = (k < K && n < N) ? W[(size_t)k * N + n] : 0.0f;
  }
  __syncthreads();
  #pragma unroll
  for (int i = 0; i < 32; i += 8){
    int n = nb + ty + i, k = kb + tx;
    if (n < N && k < K) WT[(size_t)n * K + k] = (bf16)tile[tx][ty + i];
  }
}

// ===================== rmsnorm (one token per block) =====================
__global__ __launch_bounds__(256) void rmsnorm_kernel(const float* __restrict__ x, const float* __restrict__ w,
                                                      bf16* __restrict__ outb, float* __restrict__ outf, int D){
  int t = blockIdx.x, tid = threadIdx.x;
  const float* xr = x + (size_t)t * D;
  __shared__ float red[256];
  float ss = 0.f;
  for (int d = tid; d < D; d += 256){ float v = xr[d]; ss += v * v; }
  red[tid] = ss; __syncthreads();
  for (int s = 128; s > 0; s >>= 1){ if (tid < s) red[tid] += red[tid + s]; __syncthreads(); }
  float inv = rsqrtf(red[0] / (float)D + 1e-6f);
  for (int d = tid; d < D; d += 256){
    float v = xr[d] * inv * w[d];
    if (outb) outb[(size_t)t * D + d] = (bf16)v;
    if (outf) outf[(size_t)t * D + d] = v;
  }
}

// ===================== bf16 WMMA GEMM: 128x64 tile, double-buffered async LDS =====================
// flags: 1=RELU, 2=ACCUM(Cf+=), 4=ROPE, 8=QKSPLIT(col -> (col/64)*128 + col%64 + half_off)
#define LDSK 40
__global__ __launch_bounds__(256) void gemm_bf16_wmma(
    const bf16* __restrict__ A, int lda,
    const bf16* __restrict__ Bt, int ldb,              // Bt[N][K]
    const float* __restrict__ bias,
    float* __restrict__ Cf, int ldcf,
    bf16* __restrict__ Cb, int ldcb,
    const float* __restrict__ Res, int ldr,
    int M, int N, int K, int flags, int half_off,
    const float* __restrict__ cosT, const float* __restrict__ sinT, int Sdim,
    const int* __restrict__ dynCnt, const int* __restrict__ dynOff)
{
  __shared__ __align__(16) bf16 As[2][128 * LDSK];
  __shared__ __align__(16) bf16 Bs[2][64 * LDSK];
  int m0 = blockIdx.x * 128;
  int n0 = blockIdx.y * 64;
  int mbase = 0;
  if (dynCnt){ int cnt = *dynCnt; mbase = *dynOff; if (m0 >= cnt) return; }
  int tid = threadIdx.x, w = tid >> 5, lane = tid & 31;
  int mt2 = (w & 3) * 32;          // this wave's two M-subtiles: mt2, mt2+16
  int nt  = (w >> 2) * 2;          // this wave's two N-subtiles
  v8f c00 = {}, c01 = {}, c10 = {}, c11 = {};
  // per-thread copy coordinates (16B chunks)
  int ar = tid >> 2, ac = (tid & 3) * 8;   // A rows ar, ar+64 ; B row ar
  auto issue = [&](int buf, int k0){
    cp16_g2s(&As[buf][(ar)      * LDSK + ac], &A [(size_t)(mbase + m0 + ar)      * lda + k0 + ac]);
    cp16_g2s(&As[buf][(ar + 64) * LDSK + ac], &A [(size_t)(mbase + m0 + ar + 64) * lda + k0 + ac]);
    cp16_g2s(&Bs[buf][(ar)      * LDSK + ac], &Bt[(size_t)(n0 + ar)              * ldb + k0 + ac]);
  };
  int niter = K / 32;
  issue(0, 0);
  for (int i = 0; i < niter; ++i){
    int buf = i & 1;
    if (i + 1 < niter){ issue(buf ^ 1, (i + 1) * 32); ASYNC_WAIT(3); }
    else              { ASYNC_WAIT(0); }
    __syncthreads();
    v16bf a0 = frag_a(As[buf], mt2,      LDSK, 0);
    v16bf a1 = frag_a(As[buf], mt2 + 16, LDSK, 0);
    v16bf b0 = frag_b(Bs[buf], nt * 16,      LDSK, 0);
    v16bf b1 = frag_b(Bs[buf], nt * 16 + 16, LDSK, 0);
    c00 = WMMA_BF16(a0, b0, c00);
    c01 = WMMA_BF16(a0, b1, c01);
    c10 = WMMA_BF16(a1, b0, c10);
    c11 = WMMA_BF16(a1, b1, c11);
    __syncthreads();
  }
  // epilogue: C layout VGPR r -> row +r (+8 for lanes>=16), col = subtile + lane%16
  int rhalf = (lane & 16) ? 8 : 0;
  v8f accs[2][2] = {{c00, c01}, {c10, c11}};
  #pragma unroll
  for (int mi = 0; mi < 2; ++mi){
    #pragma unroll
    for (int ni = 0; ni < 2; ++ni){
      int col = n0 + nt * 16 + ni * 16 + (lane & 15);
      float bs = bias ? bias[col] : 0.0f;
      #pragma unroll
      for (int r = 0; r < 8; ++r){
        size_t grow = (size_t)mbase + m0 + mt2 + mi * 16 + r + rhalf;
        float v = accs[mi][ni][r] + bs;
        if (flags & 2) v += Cf[grow * ldcf + col];
        if (Res)       v += Res[grow * ldr + col];
        if (flags & 1) v = fmaxf(v, 0.0f);
        if (flags & 4){
          int srow = (int)(grow % (size_t)Sdim);
          int fi = (col & 63) >> 1;
          float cv = cosT[srow * 32 + fi];
          float sv = sinT[srow * 32 + fi];
          float pv = __shfl_xor(v, 1, 32);   // RoPE pair partner (adjacent column/lane)
          v = (lane & 1) ? (pv * sv + v * cv) : (v * cv - pv * sv);
        }
        if (Cf) Cf[grow * ldcf + col] = v;
        if (Cb){
          int oc = (flags & 8) ? (((col >> 6) * 128) + (col & 63) + half_off) : col;
          Cb[grow * (size_t)ldcb + oc] = (bf16)v;
        }
      }
    }
  }
}

// ===================== flash attention (no mask), WMMA bf16, online softmax =====================
#define FD 136   // 128 + 8 pad
#define FS 72    // 64 + 8 pad
__global__ __launch_bounds__(256) void flash_attn(
    const bf16* __restrict__ qf, const bf16* __restrict__ kf, const bf16* __restrict__ vbuf,
    bf16* __restrict__ attn, int Sn)
{
  __shared__ __align__(16) bf16 qt[64 * FD];
  __shared__ __align__(16) bf16 kt[64 * FD];
  __shared__ __align__(16) bf16 vt[64 * FS];   // transposed [hd][kv]
  __shared__ __align__(16) bf16 pt[64 * FS];   // [row][kv]
  __shared__ float mrow[64], lrow[64], scrow[64];
  __shared__ float part[2][64], part2[2][64];

  int q0 = blockIdx.x * 64;
  int bh = blockIdx.y; int b = bh >> 4; int h = bh & 15;
  int tid = threadIdx.x, w = tid >> 5, lane = tid & 31;
  int mt = (w & 3) * 16;
  int p  = w >> 2;
  int nt0 = p * 2;
  int rhalf = (lane & 16) ? 8 : 0;

  for (int c = tid; c < 64 * 16; c += 256){
    int r = c >> 4, d = (c & 15) * 8;
    cp16_g2s(&qt[r * FD + d], &qf[((size_t)(b * Sn + q0 + r)) * 2048 + h * 128 + d]);
  }
  if (tid < 64){ mrow[tid] = -1e30f; lrow[tid] = 0.0f; }
  v8f o0 = {}, o1 = {};

  for (int kv0 = 0; kv0 < Sn; kv0 += 64){
    __syncthreads();
    for (int c = tid; c < 64 * 16; c += 256){
      int r = c >> 4, d = (c & 15) * 8;
      cp16_g2s(&kt[r * FD + d], &kf[((size_t)(b * Sn + kv0 + r)) * 2048 + h * 128 + d]);
    }
    for (int c = tid; c < 64 * 8; c += 256){
      int r = c >> 3, d = (c & 7) * 8;
      v8bf vv = *(const v8bf*)&vbuf[((size_t)(b * Sn + kv0 + r)) * 1024 + h * 64 + d];
      #pragma unroll
      for (int j = 0; j < 8; ++j) vt[(d + j) * FS + r] = vv[j];
    }
    ASYNC_WAIT(0);
    __syncthreads();
    // S = Q K^T / 8
    v8f s0 = {}, s1 = {};
    #pragma unroll
    for (int kc = 0; kc < 4; ++kc){
      v16bf a  = frag_a(qt, mt, FD, kc * 32);
      v16bf b0 = frag_b(kt, nt0 * 16,      FD, kc * 32);
      v16bf b1 = frag_b(kt, nt0 * 16 + 16, FD, kc * 32);
      s0 = WMMA_BF16(a, b0, s0);
      s1 = WMMA_BF16(a, b1, s1);
    }
    #pragma unroll
    for (int r = 0; r < 8; ++r){ s0[r] *= 0.125f; s1[r] *= 0.125f; }
    #pragma unroll
    for (int r = 0; r < 8; ++r){
      float val = fmaxf(s0[r], s1[r]);
      val = fmaxf(val, __shfl_xor(val, 1, 32));
      val = fmaxf(val, __shfl_xor(val, 2, 32));
      val = fmaxf(val, __shfl_xor(val, 4, 32));
      val = fmaxf(val, __shfl_xor(val, 8, 32));
      if ((lane & 15) == 0) part[p][mt + r + rhalf] = val;
    }
    __syncthreads();
    if (tid < 64){
      float mn = fmaxf(mrow[tid], fmaxf(part[0][tid], part[1][tid]));
      float sc = __expf(mrow[tid] - mn);
      scrow[tid] = sc; lrow[tid] *= sc; mrow[tid] = mn;
    }
    __syncthreads();
    #pragma unroll
    for (int r = 0; r < 8; ++r){
      int row = mt + r + rhalf;
      float mm = mrow[row];
      float p0 = __expf(s0[r] - mm), p1 = __expf(s1[r] - mm);
      pt[row * FS + nt0 * 16      + (lane & 15)] = (bf16)p0;
      pt[row * FS + nt0 * 16 + 16 + (lane & 15)] = (bf16)p1;
      float rs = p0 + p1;
      rs += __shfl_xor(rs, 1, 32);
      rs += __shfl_xor(rs, 2, 32);
      rs += __shfl_xor(rs, 4, 32);
      rs += __shfl_xor(rs, 8, 32);
      if ((lane & 15) == 0) part2[p][row] = rs;
      float sc = scrow[row];
      o0[r] *= sc; o1[r] *= sc;
    }
    __syncthreads();
    if (tid < 64) lrow[tid] += part2[0][tid] + part2[1][tid];
    __syncthreads();
    #pragma unroll
    for (int kc = 0; kc < 2; ++kc){
      v16bf a  = frag_a(pt, mt, FS, kc * 32);
      v16bf b0 = frag_b(vt, nt0 * 16,      FS, kc * 32);
      v16bf b1 = frag_b(vt, nt0 * 16 + 16, FS, kc * 32);
      o0 = WMMA_BF16(a, b0, o0);
      o1 = WMMA_BF16(a, b1, o1);
    }
  }
  __syncthreads();
  #pragma unroll
  for (int r = 0; r < 8; ++r){
    int row = mt + r + rhalf;
    float inv = 1.0f / lrow[row];
    size_t tok = (size_t)(b * Sn + q0 + row);
    attn[tok * 1024 + h * 64 + nt0 * 16      + (lane & 15)] = (bf16)(o0[r] * inv);
    attn[tok * 1024 + h * 64 + nt0 * 16 + 16 + (lane & 15)] = (bf16)(o1[r] * inv);
  }
}

// ===================== router: noisy top-2, per-token =====================
__global__ __launch_bounds__(256) void router_kernel(
    const float* __restrict__ h2, const float* __restrict__ wr, const float* __restrict__ br,
    const float* __restrict__ wn, const float* __restrict__ bn, const float* __restrict__ noise,
    int* __restrict__ idx2, float* __restrict__ wv2, int* __restrict__ cnt, int D)
{
  int t = blockIdx.x, tid = threadIdx.x;
  float pr[8] = {0}, pn[8] = {0};
  for (int d = tid; d < D; d += 256){
    float xv = h2[(size_t)t * D + d];
    const float* wrr = wr + (size_t)d * 8;
    const float* wnn = wn + (size_t)d * 8;
    #pragma unroll
    for (int e = 0; e < 8; ++e){ pr[e] += xv * wrr[e]; pn[e] += xv * wnn[e]; }
  }
  __shared__ float red[256];
  __shared__ float rsum[16];
  for (int e = 0; e < 16; ++e){
    red[tid] = (e < 8) ? pr[e] : pn[e - 8];
    __syncthreads();
    for (int s = 128; s > 0; s >>= 1){ if (tid < s) red[tid] += red[tid + s]; __syncthreads(); }
    if (tid == 0) rsum[e] = red[0];
    __syncthreads();
  }
  if (tid == 0){
    float nz[8];
    #pragma unroll
    for (int e = 0; e < 8; ++e){
      float si = rsum[8 + e] + bn[e];
      float sp = (si > 20.0f) ? si : log1pf(__expf(si));
      nz[e] = rsum[e] + br[e] + noise[(size_t)t * 8 + e] * sp;
    }
    int i0 = 0;
    for (int e = 1; e < 8; ++e) if (nz[e] > nz[i0]) i0 = e;
    int i1 = -1;
    for (int e = 0; e < 8; ++e){ if (e == i0) continue; if (i1 < 0 || nz[e] > nz[i1]) i1 = e; }
    float eb = __expf(nz[i1] - nz[i0]);
    float den = 1.0f + eb;
    idx2[t * 2 + 0] = i0; idx2[t * 2 + 1] = i1;
    wv2[t * 2 + 0] = 1.0f / den; wv2[t * 2 + 1] = eb / den;
    atomicAdd(&cnt[i0], 1); atomicAdd(&cnt[i1], 1);
  }
}

__global__ void zero8(int* p){ if (threadIdx.x < 8) p[threadIdx.x] = 0; }

__global__ void scan_kernel(const int* __restrict__ cnt, int* __restrict__ offs, int* __restrict__ pos){
  if (threadIdx.x == 0){
    int a = 0;
    for (int e = 0; e < 8; ++e){ offs[e] = a; pos[e] = a; a += cnt[e]; }
  }
}

__global__ __launch_bounds__(256) void assign_kernel(const int* __restrict__ idx2, int* __restrict__ pos,
                                                     int* __restrict__ slotOf, int* __restrict__ slotTok, int T){
  int t = blockIdx.x * 256 + threadIdx.x;
  if (t >= T) return;
  #pragma unroll
  for (int k = 0; k < 2; ++k){
    int e = idx2[t * 2 + k];
    int s = atomicAdd(&pos[e], 1);
    slotOf[t * 2 + k] = s;
    slotTok[s] = t;
  }
}

__global__ __launch_bounds__(256) void gather_kernel(const bf16* __restrict__ h2b, const int* __restrict__ slotTok,
                                                     bf16* __restrict__ rin, int D){
  int s = blockIdx.x;
  int tok = slotTok[s];
  for (int d = threadIdx.x; d < D; d += 256)
    rin[(size_t)s * D + d] = h2b[(size_t)tok * D + d];
}

__global__ __launch_bounds__(256) void combine_kernel(
    const float* __restrict__ x2, const float* __restrict__ h2, const float* __restrict__ fsh,
    const float* __restrict__ orout, const int* __restrict__ slotOf, const float* __restrict__ wv2,
    float* __restrict__ out, int D)
{
  int t = blockIdx.x;
  int s0 = slotOf[t * 2], s1 = slotOf[t * 2 + 1];
  float g0 = wv2[t * 2], g1 = wv2[t * 2 + 1];
  for (int d = threadIdx.x; d < D; d += 256){
    size_t o = (size_t)t * D + d;
    out[o] = x2[o] + h2[o] + fsh[o] + g0 * orout[(size_t)s0 * D + d] + g1 * orout[(size_t)s1 * D + d];
  }
}

// ===================== host orchestration =====================
extern "C" void kernel_launch(void* const* d_in, const int* in_sizes, int n_in,
                              void* d_out, int out_size, void* d_ws, size_t ws_size,
                              hipStream_t stream)
{
  (void)in_sizes; (void)n_in; (void)out_size; (void)ws_size;
  const float* x       = (const float*)d_in[0];
  const float* fcos    = (const float*)d_in[1];
  const float* fsin    = (const float*)d_in[2];
  const float* rnoise  = (const float*)d_in[3];
  const float* rms1w   = (const float*)d_in[4];
  const float* rms2w   = (const float*)d_in[5];
  const float* w_lq    = (const float*)d_in[6];
  const float* w_lkv   = (const float*)d_in[7];
  const float* w_q     = (const float*)d_in[8];
  const float* w_k     = (const float*)d_in[9];
  const float* w_v     = (const float*)d_in[10];
  const float* w_qr    = (const float*)d_in[11];
  const float* b_qr    = (const float*)d_in[12];
  const float* w_kr    = (const float*)d_in[13];
  const float* b_kr    = (const float*)d_in[14];
  const float* w_o     = (const float*)d_in[15];
  const float* b_o     = (const float*)d_in[16];
  const float* w_route = (const float*)d_in[17];
  const float* b_route = (const float*)d_in[18];
  const float* w_noise = (const float*)d_in[19];
  const float* b_noise = (const float*)d_in[20];
  const float* rW1     = (const float*)d_in[21];
  const float* rb1     = (const float*)d_in[22];
  const float* rW2     = (const float*)d_in[23];
  const float* rb2     = (const float*)d_in[24];
  const float* sW1     = (const float*)d_in[25];
  const float* sb1     = (const float*)d_in[26];
  const float* sW2     = (const float*)d_in[27];
  const float* sb2     = (const float*)d_in[28];
  float* out = (float*)d_out;

  char* ws = (char*)d_ws;
  size_t off = 0;
  auto alloc = [&](size_t bytes)->char*{
    char* p = ws + off;
    off += (bytes + 255) & ~(size_t)255;
    return p;
  };
  bf16* wt_lq  = (bf16*)alloc((size_t)LQQ * DD * 2);
  bf16* wt_lkv = (bf16*)alloc((size_t)LKV * DD * 2);
  bf16* wt_q   = (bf16*)alloc((size_t)1024 * LQQ * 2);
  bf16* wt_qr  = (bf16*)alloc((size_t)1024 * LQQ * 2);
  bf16* wt_k   = (bf16*)alloc((size_t)1024 * LKV * 2);
  bf16* wt_v   = (bf16*)alloc((size_t)1024 * LKV * 2);
  bf16* wt_kr  = (bf16*)alloc((size_t)1024 * DD * 2);
  bf16* wt_o   = (bf16*)alloc((size_t)DD * 1024 * 2);
  bf16* wt_s1  = (bf16*)alloc((size_t)2 * 4096 * 1024 * 2);
  bf16* wt_s2  = (bf16*)alloc((size_t)2 * 1024 * 4096 * 2);
  bf16* wt_r1  = (bf16*)alloc((size_t)8 * 4096 * 1024 * 2);
  bf16* wt_r2  = (bf16*)alloc((size_t)8 * 1024 * 4096 * 2);
  bf16* h_bf   = (bf16*)alloc((size_t)TT * DD * 2);
  bf16* cq_bf  = (bf16*)alloc((size_t)TT * LQQ * 2);
  bf16* ckv_bf = (bf16*)alloc((size_t)TT * LKV * 2);
  bf16* qf_bf  = (bf16*)alloc((size_t)TT * 2048 * 2);
  bf16* kf_bf  = (bf16*)alloc((size_t)TT * 2048 * 2);
  bf16* v_bf   = (bf16*)alloc((size_t)TT * 1024 * 2);
  bf16* at_bf  = (bf16*)alloc((size_t)TT * 1024 * 2);
  float* x2_f  = (float*)alloc((size_t)TT * DD * 4);
  float* h2_f  = (float*)alloc((size_t)TT * DD * 4);
  bf16* h2_bf  = (bf16*)alloc((size_t)TT * DD * 2);
  float* fsh_f = (float*)alloc((size_t)TT * DD * 4);
  bf16* hid_bf = (bf16*)alloc((size_t)SLOTPAD * 4096 * 2);
  bf16* rin_bf = (bf16*)alloc((size_t)SLOTPAD * DD * 2);
  float* oro_f = (float*)alloc((size_t)SLOTPAD * DD * 4);
  int*  idx2    = (int*)alloc(NSLOT * 4);
  float* wv2    = (float*)alloc(NSLOT * 4);
  int*  slotOf  = (int*)alloc(NSLOT * 4);
  int*  slotTok = (int*)alloc(SLOTPAD * 4);
  int*  cnt     = (int*)alloc(8 * 4);
  int*  offs    = (int*)alloc(8 * 4);
  int*  pos     = (int*)alloc(8 * 4);

  auto tr = [&](const float* W, bf16* WT, int K, int N){
    dim3 g(N / 32, K / 32);
    transpose_cvt<<<g, 256, 0, stream>>>(W, WT, K, N);
  };
  auto gemm = [&](const bf16* A, int lda, const bf16* Bt, int ldb, const float* bias,
                  float* Cf, int ldcf, bf16* Cb, int ldcb, const float* Res, int ldr,
                  int M, int N, int K, int flags, int half_off,
                  const int* dynCnt, const int* dynOff){
    dim3 g(M / 128, N / 64);
    gemm_bf16_wmma<<<g, 256, 0, stream>>>(A, lda, Bt, ldb, bias, Cf, ldcf, Cb, ldcb, Res, ldr,
                                          M, N, K, flags, half_off, fcos, fsin, SS, dynCnt, dynOff);
  };

  // ---- weight convert + transpose ----
  tr(w_lq,  wt_lq, 1024, 768);
  tr(w_lkv, wt_lkv, 1024, 512);
  tr(w_q,   wt_q, 768, 1024);
  tr(w_qr,  wt_qr, 768, 1024);
  tr(w_k,   wt_k, 512, 1024);
  tr(w_v,   wt_v, 512, 1024);
  tr(w_kr,  wt_kr, 1024, 1024);
  tr(w_o,   wt_o, 1024, 1024);
  for (int j = 0; j < 2; ++j){
    tr(sW1 + (size_t)j * 1024 * 4096, wt_s1 + (size_t)j * 4096 * 1024, 1024, 4096);
    tr(sW2 + (size_t)j * 4096 * 1024, wt_s2 + (size_t)j * 1024 * 4096, 4096, 1024);
  }
  for (int e = 0; e < 8; ++e){
    tr(rW1 + (size_t)e * 1024 * 4096, wt_r1 + (size_t)e * 4096 * 1024, 1024, 4096);
    tr(rW2 + (size_t)e * 4096 * 1024, wt_r2 + (size_t)e * 1024 * 4096, 4096, 1024);
  }

  // ---- attention path ----
  rmsnorm_kernel<<<TT, 256, 0, stream>>>(x, rms1w, h_bf, nullptr, DD);
  gemm(h_bf, DD,  wt_lq, DD,  nullptr, nullptr,0, cq_bf, LQQ, nullptr,0, TT, LQQ, DD, 0,0, nullptr,nullptr);
  gemm(h_bf, DD,  wt_lkv,DD,  nullptr, nullptr,0, ckv_bf,LKV, nullptr,0, TT, LKV, DD, 0,0, nullptr,nullptr);
  gemm(cq_bf, LQQ, wt_q,  LQQ, nullptr, nullptr,0, qf_bf, 2048, nullptr,0, TT, 1024, LQQ, 8, 0,  nullptr,nullptr);
  gemm(cq_bf, LQQ, wt_qr, LQQ, b_qr,    nullptr,0, qf_bf, 2048, nullptr,0, TT, 1024, LQQ, 8|4, 64, nullptr,nullptr);
  gemm(ckv_bf,LKV, wt_k,  LKV, nullptr, nullptr,0, kf_bf, 2048, nullptr,0, TT, 1024, LKV, 8, 0,  nullptr,nullptr);
  gemm(h_bf,  DD,  wt_kr, DD,  b_kr,    nullptr,0, kf_bf, 2048, nullptr,0, TT, 1024, DD,  8|4, 64, nullptr,nullptr);
  gemm(ckv_bf,LKV, wt_v,  LKV, nullptr, nullptr,0, v_bf,  1024, nullptr,0, TT, 1024, LKV, 0, 0,  nullptr,nullptr);
  {
    dim3 g(SS / 64, BB * HH);
    flash_attn<<<g, 256, 0, stream>>>(qf_bf, kf_bf, v_bf, at_bf, SS);
  }
  gemm(at_bf, 1024, wt_o, 1024, b_o, x2_f, DD, nullptr, 0, x, DD, TT, DD, 1024, 0, 0, nullptr, nullptr);

  // ---- MoE path ----
  rmsnorm_kernel<<<TT, 256, 0, stream>>>(x2_f, rms2w, h2_bf, h2_f, DD);
  zero8<<<1, 32, 0, stream>>>(cnt);
  router_kernel<<<TT, 256, 0, stream>>>(h2_f, w_route, b_route, w_noise, b_noise, rnoise, idx2, wv2, cnt, DD);
  scan_kernel<<<1, 32, 0, stream>>>(cnt, offs, pos);
  assign_kernel<<<TT / 256, 256, 0, stream>>>(idx2, pos, slotOf, slotTok, TT);
  gather_kernel<<<NSLOT, 256, 0, stream>>>(h2_bf, slotTok, rin_bf, DD);

  for (int j = 0; j < 2; ++j){
    gemm(h2_bf, DD, wt_s1 + (size_t)j * 4096 * 1024, DD, sb1 + (size_t)j * 4096,
         nullptr,0, hid_bf, 4096, nullptr,0, TT, 4096, DD, 1, 0, nullptr, nullptr);
    gemm(hid_bf, 4096, wt_s2 + (size_t)j * 1024 * 4096, 4096, sb2 + (size_t)j * 1024,
         fsh_f, DD, nullptr,0, nullptr,0, TT, DD, 4096, (j == 0) ? 0 : 2, 0, nullptr, nullptr);
  }
  for (int e = 0; e < 8; ++e)
    gemm(rin_bf, DD, wt_r1 + (size_t)e * 4096 * 1024, DD, rb1 + (size_t)e * 4096,
         nullptr,0, hid_bf, 4096, nullptr,0, TT, 4096, DD, 1, 0, cnt + e, offs + e);
  for (int e = 0; e < 8; ++e)
    gemm(hid_bf, 4096, wt_r2 + (size_t)e * 1024 * 4096, 4096, rb2 + (size_t)e * 1024,
         oro_f, DD, nullptr,0, nullptr,0, TT, DD, 4096, 0, 0, cnt + e, offs + e);

  combine_kernel<<<TT, 256, 0, stream>>>(x2_f, h2_f, fsh_f, oro_f, slotOf, wv2, out, DD);
}
